// ClusterForecasting_47304769798389
// MI455X (gfx1250) — matrix operand order repaired and checked by reading, not compiled, not run on hardware
//
#include <hip/hip_runtime.h>
#include <hip/hip_bf16.h>
#include <stdint.h>

#define B_  64
#define S_  512
#define F_  8
#define D_  512
#define H_  8
#define L_  2
#define K_  512
#define FF_ 2048
#define DH_ 64
#define M_  (B_*S_)   // 32768 rows of the token stream

typedef unsigned short u16;
typedef __attribute__((ext_vector_type(16))) __bf16       v16bf;
typedef __attribute__((ext_vector_type(8)))  float        v8f;
typedef __attribute__((ext_vector_type(4)))  float        v4f;
typedef __attribute__((ext_vector_type(4)))  unsigned int v4u;

union Frag { v4u q[2]; v16bf v; };

__device__ __forceinline__ u16 f2bf(float f) {
    unsigned int u = __float_as_uint(f);
    u += 0x7fffu + ((u >> 16) & 1u);          // round-to-nearest-even
    return (u16)(u >> 16);
}

__device__ __forceinline__ v8f wmma_bf16(v16bf a, v16bf b, v8f c) {
    return __builtin_amdgcn_wmma_f32_16x16x32_bf16(
        /*neg_a=*/false, a, /*neg_b=*/false, b,
        /*c_mod=*/(short)0, c, /*reuse_a=*/false, /*reuse_b=*/false);
}

// ---------------------------------------------------------------------------
// h[row, :] = x[row, :8] @ enc_W + enc_b          (K=8, plain VALU)
// ---------------------------------------------------------------------------
__global__ void encode_k(const float* __restrict__ x, const float* __restrict__ W,
                         const float* __restrict__ bvec, float* __restrict__ h) {
    int row = blockIdx.x, tid = threadIdx.x;
    float xr[8];
#pragma unroll
    for (int j = 0; j < 8; ++j) xr[j] = x[(size_t)row * 8 + j];
#pragma unroll
    for (int cc = 0; cc < 2; ++cc) {
        int col = tid + cc * 256;
        float acc = bvec[col];
#pragma unroll
        for (int j = 0; j < 8; ++j) acc += xr[j] * W[j * 512 + col];
        h[(size_t)row * 512 + col] = acc;
    }
}

// ---------------------------------------------------------------------------
// Row LayerNorm (f32 in) -> bf16 out. One 256-thread block per row, D=512.
// ---------------------------------------------------------------------------
__global__ void ln_bf16(const float* __restrict__ h, u16* __restrict__ out) {
    __shared__ float ssum[256], ssq[256];
    int row = blockIdx.x, tid = threadIdx.x;
    const float* hr = h + (size_t)row * D_;
    float a = hr[tid], c = hr[tid + 256];
    ssum[tid] = a + c;
    ssq[tid]  = a * a + c * c;
    __syncthreads();
    for (int st = 128; st > 0; st >>= 1) {
        if (tid < st) { ssum[tid] += ssum[tid + st]; ssq[tid] += ssq[tid + st]; }
        __syncthreads();
    }
    float mean = ssum[0] * (1.f / D_);
    float var  = ssq[0] * (1.f / D_) - mean * mean;
    float rstd = rsqrtf(var + 1e-5f);
    u16* orow = out + (size_t)row * D_;
    orow[tid]       = f2bf((a - mean) * rstd);
    orow[tid + 256] = f2bf((c - mean) * rstd);
}

// ---------------------------------------------------------------------------
// Pack f32 weight matrix [Kdim x N] (optionally accessed transposed) into the
// WMMA B-fragment layout: P[((kt*nTiles+nt)*32 + lane)*16 + e] where
// k = kt*32 + (lane>>4)*16 + e, n = nt*16 + (lane&15).
// ---------------------------------------------------------------------------
__global__ void pack_b(const float* __restrict__ W, u16* __restrict__ P,
                       int Kdim, int N, int ld, int transpose) {
    int t = blockIdx.x * blockDim.x + threadIdx.x;
    int total = (Kdim >> 5) * (N >> 4) * 32;
    if (t >= total) return;
    int lane = t & 31, tile = t >> 5;
    int nTiles = N >> 4;
    int kt = tile / nTiles, nt = tile % nTiles;
    int hl = lane >> 4, ln = lane & 15;
    int n = nt * 16 + ln;
#pragma unroll
    for (int e = 0; e < 16; ++e) {
        int k = kt * 32 + hl * 16 + e;
        float v = transpose ? W[(size_t)n * ld + k] : W[(size_t)k * ld + n];
        P[(size_t)t * 16 + e] = f2bf(v);
    }
}

// ---------------------------------------------------------------------------
// Pack V (from bf16 qkv buffer) into B-fragment layout per (b,h):
// vp[(((bh*16+kb)*4+j)*32+lane)*16+e] = V[key = kb*32+(lane>>4)*16+e][dh = j*16+(lane&15)]
// ---------------------------------------------------------------------------
__global__ void pack_v_kernel(const u16* __restrict__ qkv, u16* __restrict__ vp) {
    int t = blockIdx.x * blockDim.x + threadIdx.x;
    if (t >= B_ * H_ * 16 * 4 * 32) return;
    int lane = t & 31, rest = t >> 5;
    int j  = rest & 3;
    int kb = (rest >> 2) & 15;
    int bh = rest >> 6;
    int b = bh >> 3, h = bh & 7;
    int hl = lane >> 4, ln = lane & 15;
    int n = j * 16 + ln;
#pragma unroll
    for (int e = 0; e < 16; ++e) {
        int k = kb * 32 + hl * 16 + e;
        vp[(size_t)t * 16 + e] = qkv[((size_t)(b * S_ + k)) * 1536 + 1024 + h * 64 + n];
    }
}

// ---------------------------------------------------------------------------
// GEMM: out[M,N] = A_bf16[M,Kdim] @ Bpacked (+bias, relu, residual).
// One wave per block; wave tile = 32(M) x 64(N); K-steps of 32.
// ---------------------------------------------------------------------------
template<bool BIAS, bool RELU, bool RES, bool OUTBF16>
__global__ void gemm_bf16(const u16* __restrict__ A, const u16* __restrict__ Bp,
                          int N, int Kdim,
                          const float* __restrict__ bias,
                          const float* __restrict__ res,
                          float* __restrict__ outF, u16* __restrict__ outH) {
    const int lane = threadIdx.x;
    const int hl = lane >> 4, ln = lane & 15;
    const int nt0 = blockIdx.x * 4;
    const int m0  = blockIdx.y * 32;
    const int nTiles = N >> 4;
    const int kTiles = Kdim >> 5;

    v8f acc[2][4] = {};
    for (int kt = 0; kt < kTiles; ++kt) {
        Frag a[2];
#pragma unroll
        for (int mi = 0; mi < 2; ++mi) {
            const u16* ap = A + (size_t)(m0 + mi * 16 + ln) * Kdim + kt * 32;
            a[mi].q[0] = *(const v4u*)(ap + hl * 16);
            a[mi].q[1] = *(const v4u*)(ap + hl * 16 + 8);
        }
#pragma unroll
        for (int nj = 0; nj < 4; ++nj) {
            Frag b;
            const u16* bp = Bp + ((size_t)((size_t)kt * nTiles + nt0 + nj) * 32 + lane) * 16;
            b.q[0] = *(const v4u*)(bp);
            b.q[1] = *(const v4u*)(bp + 8);
            acc[0][nj] = wmma_bf16(a[0].v, b.v, acc[0][nj]);
            acc[1][nj] = wmma_bf16(a[1].v, b.v, acc[1][nj]);
        }
    }
#pragma unroll
    for (int mi = 0; mi < 2; ++mi)
#pragma unroll
        for (int nj = 0; nj < 4; ++nj)
#pragma unroll
            for (int i = 0; i < 8; ++i) {
                int row = m0 + mi * 16 + i + 8 * hl;    // C layout: row = i + 8*(lane/16)
                int col = (nt0 + nj) * 16 + ln;         //           col = lane%16
                float v = acc[mi][nj][i];
                if (BIAS) v += bias[col];
                if (RELU) v = fmaxf(v, 0.f);
                if (RES)  v += res[(size_t)row * N + col];
                if (OUTBF16) outH[(size_t)row * N + col] = f2bf(v);
                else         outF[(size_t)row * N + col] = v;
            }
}

// ---------------------------------------------------------------------------
// Flash attention: one wave per (b,h,q-tile of 16). DH=64, S=512.
// Softmax uses a row-owner LDS pass (1 shuffle per reduction) instead of
// per-element ds_bpermute butterflies.
// ---------------------------------------------------------------------------
__global__ void flash_attn(const u16* __restrict__ qkv, const u16* __restrict__ vp,
                           u16* __restrict__ o) {
    __shared__ __align__(16) float sS[16 * 32];   // score tile, f32
    __shared__ __align__(16) u16   sP[16 * 32];   // P tile, bf16, A-layout (row-major)
    __shared__ __align__(16) float sAl[16];       // per-row alpha
    __shared__ __align__(16) float sLi[16];       // per-row 1/l
    const int lane = threadIdx.x;
    const int hl = lane >> 4, ln = lane & 15;
    const int bh = blockIdx.x;
    const int b = bh >> 3, h = bh & 7;
    const int qt = blockIdx.y;

    Frag qf[2];
    {
        const u16* qp = qkv + ((size_t)(b * S_ + qt * 16 + ln)) * 1536 + h * 64;
#pragma unroll
        for (int dt = 0; dt < 2; ++dt) {
            qf[dt].q[0] = *(const v4u*)(qp + dt * 32 + hl * 16);
            qf[dt].q[1] = *(const v4u*)(qp + dt * 32 + hl * 16 + 8);
        }
    }

    float m_own = -1e30f;   // row = ln (duplicated across the two 16-lane halves)
    float l_own = 0.f;
    v8f oacc[4] = {};

    for (int kb = 0; kb < 16; ++kb) {          // 16 key-blocks of 32
        v8f s[2] = {};
#pragma unroll
        for (int nt = 0; nt < 2; ++nt) {
            const u16* kp = qkv + ((size_t)(b * S_ + kb * 32 + nt * 16 + ln)) * 1536
                          + 512 + h * 64 + hl * 16;
#pragma unroll
            for (int dt = 0; dt < 2; ++dt) {
                Frag kf;
                kf.q[0] = *(const v4u*)(kp + dt * 32);
                kf.q[1] = *(const v4u*)(kp + dt * 32 + 8);
                s[nt] = wmma_bf16(qf[dt].v, kf.v, s[nt]);
            }
        }
        // spill the 16x32 score tile (C layout: row = i+8*hl, cols ln / 16+ln)
#pragma unroll
        for (int i = 0; i < 8; ++i) {
            sS[(i + 8 * hl) * 32 + ln]      = s[0][i];
            sS[(i + 8 * hl) * 32 + 16 + ln] = s[1][i];
        }
        __syncthreads();
        // row-owner pass: lane owns row ln, half-columns hl*16..hl*16+15
        {
            float rv[16];
            const v4f* rp = (const v4f*)&sS[ln * 32 + hl * 16];
            v4f rr0 = rp[0], rr1 = rp[1], rr2 = rp[2], rr3 = rp[3];
#pragma unroll
            for (int t = 0; t < 4; ++t) {
                rv[t]      = rr0[t] * 0.125f;   // 1/sqrt(64)
                rv[4 + t]  = rr1[t] * 0.125f;
                rv[8 + t]  = rr2[t] * 0.125f;
                rv[12 + t] = rr3[t] * 0.125f;
            }
            float mx = rv[0];
#pragma unroll
            for (int e = 1; e < 16; ++e) mx = fmaxf(mx, rv[e]);
            mx = fmaxf(mx, __shfl_xor(mx, 16, 32));       // combine the two half-rows
            float mnew  = fmaxf(m_own, mx);
            float alpha = __expf(m_own - mnew);
            float p[16], ssm = 0.f;
#pragma unroll
            for (int e = 0; e < 16; ++e) { p[e] = __expf(rv[e] - mnew); ssm += p[e]; }
            ssm += __shfl_xor(ssm, 16, 32);
            l_own = l_own * alpha + ssm;
            m_own = mnew;
            // write P (bf16) into row-major tile = A-fragment source layout
            v4u pk0, pk1;
#pragma unroll
            for (int t = 0; t < 4; ++t) {
                pk0[t] = (unsigned)f2bf(p[2 * t])     | ((unsigned)f2bf(p[2 * t + 1]) << 16);
                pk1[t] = (unsigned)f2bf(p[8 + 2 * t]) | ((unsigned)f2bf(p[9 + 2 * t]) << 16);
            }
            v4u* pw = (v4u*)&sP[ln * 32 + hl * 16];
            pw[0] = pk0;
            pw[1] = pk1;
            if (hl == 0) sAl[ln] = alpha;
        }
        __syncthreads();
        // rescale O accumulators (rows 8*hl..8*hl+7 are contiguous in sAl)
        {
            const v4f* ap_ = (const v4f*)&sAl[8 * hl];
            v4f a0 = ap_[0], a1 = ap_[1];
            float a8[8];
#pragma unroll
            for (int t = 0; t < 4; ++t) { a8[t] = a0[t]; a8[4 + t] = a1[t]; }
#pragma unroll
            for (int j = 0; j < 4; ++j)
#pragma unroll
                for (int i = 0; i < 8; ++i) oacc[j][i] *= a8[i];
        }
        Frag pf;   // P as an A-fragment (16x32)
        pf.q[0] = *(const v4u*)(&sP[ln * 32 + hl * 16]);
        pf.q[1] = *(const v4u*)(&sP[ln * 32 + hl * 16 + 8]);
#pragma unroll
        for (int j = 0; j < 4; ++j) {
            Frag vf;
            const u16* vpp = vp + ((size_t)(((bh * 16 + kb) * 4 + j) * 32 + lane)) * 16;
            vf.q[0] = *(const v4u*)(vpp);
            vf.q[1] = *(const v4u*)(vpp + 8);
            oacc[j] = wmma_bf16(pf.v, vf.v, oacc[j]);
        }
        __syncthreads();   // protect sS/sP/sAl before next iteration overwrites
    }
    if (hl == 0) sLi[ln] = 1.f / l_own;
    __syncthreads();
    {
        const v4f* lp = (const v4f*)&sLi[8 * hl];
        v4f l0 = lp[0], l1 = lp[1];
        float li8[8];
#pragma unroll
        for (int t = 0; t < 4; ++t) { li8[t] = l0[t]; li8[4 + t] = l1[t]; }
#pragma unroll
        for (int i = 0; i < 8; ++i) {
            int row = b * S_ + qt * 16 + i + 8 * hl;
#pragma unroll
            for (int j = 0; j < 4; ++j)
                o[(size_t)row * D_ + h * 64 + j * 16 + ln] = f2bf(oacc[j][i] * li8[i]);
        }
    }
}

// ---------------------------------------------------------------------------
// Misc small kernels
// ---------------------------------------------------------------------------
__global__ void f32_to_bf16(const float* __restrict__ src, u16* __restrict__ dst, int n) {
    int i = blockIdx.x * blockDim.x + threadIdx.x;
    if (i < n) dst[i] = f2bf(src[i]);
}

__global__ void centers_norm(const float* __restrict__ c, float* __restrict__ cn2) {
    int k = blockIdx.x * blockDim.x + threadIdx.x;
    if (k < K_) {
        float s = 0.f;
        for (int d = 0; d < D_; ++d) { float v = c[(size_t)k * D_ + d]; s += v * v; }
        cn2[k] = s;
    }
}

__global__ void zero_one(float* p) { if (threadIdx.x == 0 && blockIdx.x == 0) p[0] = 0.f; }

__global__ void assign_loss(const float* __restrict__ g, const float* __restrict__ cn2,
                            const float* __restrict__ hfin, const float* __restrict__ centers,
                            float* __restrict__ loss) {
    __shared__ float sred[256];
    int m = blockIdx.x * blockDim.x + threadIdx.x;   // one token row per thread
    float best = 3.4e38f; int bid = 0;
    const float* gr = g + (size_t)m * K_;
    for (int k = 0; k < K_; ++k) {
        float sc = cn2[k] - 2.f * gr[k];             // on2 is row-constant -> drop for argmin
        if (sc < best) { best = sc; bid = k; }
    }
    const float* hr = hfin + (size_t)m * D_;
    const float* cr = centers + (size_t)bid * D_;
    float acc = 0.f;
    for (int d = 0; d < D_; ++d) { float df = hr[d] - cr[d]; acc += df * df; }
    sred[threadIdx.x] = acc;
    __syncthreads();
    for (int st = 128; st > 0; st >>= 1) {
        if (threadIdx.x < st) sred[threadIdx.x] += sred[threadIdx.x + st];
        __syncthreads();
    }
    if (threadIdx.x == 0) atomicAdd(loss, sred[0] * (1.f / (B_ * D_)));
}

// ---------------------------------------------------------------------------
extern "C" void kernel_launch(void* const* d_in, const int* in_sizes, int n_in,
                              void* d_out, int out_size, void* d_ws, size_t ws_size,
                              hipStream_t stream) {
    (void)in_sizes; (void)n_in; (void)out_size; (void)ws_size;
    const float* x    = (const float*)d_in[0];
    const float* encW = (const float*)d_in[1];
    const float* encB = (const float*)d_in[2];
    const float* Wqkv = (const float*)d_in[3];
    const float* Wo   = (const float*)d_in[4];
    const float* W1   = (const float*)d_in[5];
    const float* b1   = (const float*)d_in[6];
    const float* W2   = (const float*)d_in[7];
    const float* b2   = (const float*)d_in[8];
    const float* cent = (const float*)d_in[9];
    float* out = (float*)d_out;

    char* ws = (char*)d_ws;
    size_t off = 0;
    auto take = [&](size_t bytes) -> char* {
        char* p = ws + off; off = (off + bytes + 255) & ~(size_t)255; return p;
    };
    float* h     = (float*)take((size_t)M_ * D_ * 4);          // 64 MB residual stream
    u16*   hn    = (u16*)  take((size_t)M_ * D_ * 2);          // 32 MB LN/bf16 staging
    u16*   WqkvP = (u16*)  take((size_t)L_ * D_ * 3 * D_ * 2);
    u16*   WoP   = (u16*)  take((size_t)L_ * D_ * D_ * 2);
    u16*   W1P   = (u16*)  take((size_t)L_ * D_ * FF_ * 2);
    u16*   W2P   = (u16*)  take((size_t)L_ * FF_ * D_ * 2);
    u16*   centP = (u16*)  take((size_t)D_ * K_ * 2);
    float* cn2   = (float*)take((size_t)K_ * 4);
    char*  big   = take((size_t)168 * 1024 * 1024);            // shared scratch region
    u16*   qkvB  = (u16*)big;                                                  // 96 MB
    u16*   vpack = (u16*)(big + (size_t)M_ * 3 * D_ * 2);                      // 32 MB
    u16*   obuf  = (u16*)(big + (size_t)M_ * 3 * D_ * 2
                              + (size_t)B_ * H_ * S_ * DH_ * 2);               // 32 MB
    u16*   rbuf  = (u16*)big;     // 128 MB, aliases qkv+vpack (dead by FFN)
    float* g     = (float*)big;   //  64 MB, aliases (dead by cdist)

    // ---- one-time weight packing (bf16 B-fragment layout) ----
    for (int l = 0; l < L_; ++l) {
        int tq = (D_ / 32) * (3 * D_ / 16) * 32;
        pack_b<<<(tq + 255) / 256, 256, 0, stream>>>(
            Wqkv + (size_t)l * D_ * 3 * D_, WqkvP + (size_t)l * D_ * 3 * D_, D_, 3 * D_, 3 * D_, 0);
        int to = (D_ / 32) * (D_ / 16) * 32;
        pack_b<<<(to + 255) / 256, 256, 0, stream>>>(
            Wo + (size_t)l * D_ * D_, WoP + (size_t)l * D_ * D_, D_, D_, D_, 0);
        int t1 = (D_ / 32) * (FF_ / 16) * 32;
        pack_b<<<(t1 + 255) / 256, 256, 0, stream>>>(
            W1 + (size_t)l * D_ * FF_, W1P + (size_t)l * D_ * FF_, D_, FF_, FF_, 0);
        int t2 = (FF_ / 32) * (D_ / 16) * 32;
        pack_b<<<(t2 + 255) / 256, 256, 0, stream>>>(
            W2 + (size_t)l * FF_ * D_, W2P + (size_t)l * FF_ * D_, FF_, D_, D_, 0);
    }
    int tc = (D_ / 32) * (K_ / 16) * 32;
    pack_b<<<(tc + 255) / 256, 256, 0, stream>>>(cent, centP, D_, K_, D_, 1);  // centers^T
    centers_norm<<<2, 256, 0, stream>>>(cent, cn2);

    // ---- embedding ----
    encode_k<<<M_, 256, 0, stream>>>(x, encW, encB, h);

    // ---- transformer layers ----
    for (int l = 0; l < L_; ++l) {
        ln_bf16<<<M_, 256, 0, stream>>>(h, hn);
        gemm_bf16<false, false, false, true><<<dim3((3 * D_) / 64, M_ / 32), 32, 0, stream>>>(
            hn, WqkvP + (size_t)l * D_ * 3 * D_, 3 * D_, D_, nullptr, nullptr, nullptr, qkvB);
        pack_v_kernel<<<(B_ * H_ * 16 * 4 * 32) / 256, 256, 0, stream>>>(qkvB, vpack);
        flash_attn<<<dim3(B_ * H_, S_ / 16), 32, 0, stream>>>(qkvB, vpack, obuf);
        gemm_bf16<false, false, true, false><<<dim3(D_ / 64, M_ / 32), 32, 0, stream>>>(
            obuf, WoP + (size_t)l * D_ * D_, D_, D_, nullptr, h, h, nullptr);
        ln_bf16<<<M_, 256, 0, stream>>>(h, hn);
        gemm_bf16<true, true, false, true><<<dim3(FF_ / 64, M_ / 32), 32, 0, stream>>>(
            hn, W1P + (size_t)l * D_ * FF_, FF_, D_, b1 + (size_t)l * FF_, nullptr, nullptr, rbuf);
        gemm_bf16<true, false, true, false><<<dim3(D_ / 64, M_ / 32), 32, 0, stream>>>(
            rbuf, W2P + (size_t)l * FF_ * D_, D_, FF_, b2 + (size_t)l * D_, h, h, nullptr);
    }

    // ---- outputs: output_seq then clustering loss ----
    hipMemcpyAsync(out + 1, h, (size_t)M_ * D_ * 4, hipMemcpyDeviceToDevice, stream);
    f32_to_bf16<<<(M_ * D_) / 256, 256, 0, stream>>>(h, hn, M_ * D_);
    gemm_bf16<false, false, false, false><<<dim3(K_ / 64, M_ / 32), 32, 0, stream>>>(
        hn, centP, K_, D_, nullptr, nullptr, g, nullptr);
    zero_one<<<1, 1, 0, stream>>>(out);
    assign_loss<<<M_ / 256, 256, 0, stream>>>(g, cn2, h, cent, out);
}
